// DiffLinearAttentionWeights_31920196944162
// MI455X (gfx1250) — matrix operand order repaired and checked by reading, not compiled, not run on hardware
//
#include <hip/hip_runtime.h>
#include <math.h>

typedef __attribute__((ext_vector_type(16))) _Float16 v16h;
typedef __attribute__((ext_vector_type(8)))  _Float16 v8h;
typedef __attribute__((ext_vector_type(8)))  float    v8f;
typedef __attribute__((ext_vector_type(4)))  float    f4;

#define NB  4
#define NH  16
#define NT  1024
#define ND  64
#define NBH 64                      // B*H
#define LAMBDA_INIT 0.7836057666f   // 0.8 - 0.6*exp(-3.6)

static __device__ __forceinline__ v16h cat8(v8h lo, v8h hi) {
    return __builtin_shufflevector(lo, hi, 0, 1, 2, 3, 4, 5, 6, 7,
                                           8, 9, 10, 11, 12, 13, 14, 15);
}

// ---------------------------------------------------------------------------
// Kernel 0: transpose + f16-convert weights: Wt[which][h][e][d] = W[h][d][e]
// grid = 64 (which*16 + h), block = 256
// ---------------------------------------------------------------------------
__global__ void wtr_kernel(const float* __restrict__ W1q, const float* __restrict__ W1k,
                           const float* __restrict__ W2q, const float* __restrict__ W2k,
                           _Float16* __restrict__ Wt) {
    const int which = blockIdx.x >> 4;
    const int h     = blockIdx.x & 15;
    const float* W = (which == 0) ? W1q : (which == 1) ? W1k : (which == 2) ? W2q : W2k;
    W += (size_t)h * ND * ND;
    _Float16* O = Wt + (size_t)blockIdx.x * ND * ND;

    __shared__ _Float16 s[ND * 65];            // padded stride 65 vs bank conflicts
    for (int i = threadIdx.x; i < ND * ND; i += blockDim.x) {
        const int d = i >> 6, e = i & 63;      // coalesced read of W[d][e]
        s[e * 65 + d] = (_Float16)W[i];
    }
    __syncthreads();
    for (int i = threadIdx.x; i < ND * ND; i += blockDim.x) {
        const int e = i >> 6, d = i & 63;      // coalesced write of Wt[e][d]
        O[i] = s[e * 65 + d];
    }
}

// ---------------------------------------------------------------------------
// Kernel 1: lambda_full = exp(sum(lq1*lk1)) - exp(sum(lq2*lk2)) + LAMBDA_INIT
// ---------------------------------------------------------------------------
__global__ void lambda_kernel(const float* lq1, const float* lk1,
                              const float* lq2, const float* lk2, float* out) {
    __shared__ float s1[ND], s2[ND];
    int t = threadIdx.x;                       // 64 threads
    s1[t] = lq1[t] * lk1[t];
    s2[t] = lq2[t] * lk2[t];
    __syncthreads();
    if (t == 0) {
        float a = 0.f, b = 0.f;
        for (int i = 0; i < ND; ++i) { a += s1[i]; b += s2[i]; }
        out[0] = __expf(a) - __expf(b) + LAMBDA_INIT;
    }
}

// ---------------------------------------------------------------------------
// Kernel 2: feature maps  feat = X[bh] @ W[h]   ([T,64]@[64,64]) -> f16
// grid = (64 bh, 16 rowblocks of 64 rows, 4 which), block = 128 (4 waves)
// which: 0:q@W1q->qf1  1:k@W1k->kf1  2:q@W2q->qf2  3:k@W2k->kf2
// B operand comes from pre-transposed f16 Wt (contiguous per-lane 32B loads).
// ---------------------------------------------------------------------------
__global__ void feat_kernel(const float* __restrict__ q, const float* __restrict__ k,
                            const _Float16* __restrict__ Wt,
                            _Float16* __restrict__ qf1, _Float16* __restrict__ kf1,
                            _Float16* __restrict__ qf2, _Float16* __restrict__ kf2) {
    const int bh = blockIdx.x;
    const int h  = bh & (NH - 1);
    const int rb = blockIdx.y;                 // 64-row block
    const int which = blockIdx.z;

    const float* X = (which == 0 || which == 2) ? q : k;
    _Float16* O = (which == 0) ? qf1 : (which == 1) ? kf1 : (which == 2) ? qf2 : kf2;
    X += (size_t)bh * NT * ND;
    O += (size_t)bh * NT * ND;
    const _Float16* Wh = Wt + (size_t)(which * NH + h) * ND * ND;  // [e][d] f16

    const int wave = threadIdx.x >> 5;
    const int lane = threadIdx.x & 31;
    const int m    = lane & 15;
    const int hg   = lane >> 4;
    const int rowbase = rb * 64 + wave * 16;

    // A tiles: rows of X (f32 -> f16); lane = row m,
    // contiguous [kb+16*hg, +16)
    v16h a[2];
    const float* xr = X + (size_t)(rowbase + m) * ND;
#pragma unroll
    for (int ki = 0; ki < 2; ++ki) {
        const int kb = ki * 32;
        const f4 p0 = *(const f4*)(xr + kb + 16 * hg);
        const f4 p1 = *(const f4*)(xr + kb + 16 * hg + 4);
        const f4 p2 = *(const f4*)(xr + kb + 16 * hg + 8);
        const f4 p3 = *(const f4*)(xr + kb + 16 * hg + 12);
#pragma unroll
        for (int t2 = 0; t2 < 4; ++t2) {
            a[ki][t2]      = (_Float16)p0[t2];
            a[ki][4 + t2]  = (_Float16)p1[t2];
            a[ki][8 + t2]  = (_Float16)p2[t2];
            a[ki][12 + t2] = (_Float16)p3[t2];
        }
    }

    // 4 column tiles of 16
#pragma unroll
    for (int nt = 0; nt < 4; ++nt) {
        v8f c = {};
        const int e = nt * 16 + m;             // output column (lane -> N)
        const _Float16* wcol = Wh + (size_t)e * ND + 16 * hg;
#pragma unroll
        for (int ki = 0; ki < 2; ++ki) {
            const v16h b = *(const v16h*)(wcol + ki * 32);  // 16 contiguous K halves
            c = __builtin_amdgcn_wmma_f32_16x16x32_f16(false, a[ki], false, b,
                                                       (short)0, c, false, false);
        }
        // C layout: VGPR r, lane -> row = r + 8*hg, col = e
#pragma unroll
        for (int r = 0; r < 8; ++r)
            O[(size_t)(rowbase + r + 8 * hg) * ND + e] = (_Float16)c[r];
    }
}

// ---------------------------------------------------------------------------
// Kernel 3: causal scores + normalize + differential combine, fused.
// grid.x = 64*64 (bh * 16-row tile), block = 256 (8 waves)
// dynamic LDS: aw1[16][1024], aw2[16][1024], partial[256], sinv[32]
// ---------------------------------------------------------------------------
__global__ void attn_kernel(const _Float16* __restrict__ qf1, const _Float16* __restrict__ kf1,
                            const _Float16* __restrict__ qf2, const _Float16* __restrict__ kf2,
                            const float* __restrict__ lambdaPtr, float* __restrict__ out) {
    extern __shared__ float smem[];
    float* aw1     = smem;                       // 16*1024
    float* aw2     = smem + 16 * NT;             // 16*1024
    float* partial = smem + 32 * NT;             // 256
    float* sinv    = partial + 256;              // 32

    const int bh = blockIdx.x >> 6;
    const int it = blockIdx.x & 63;              // row-tile index (rows 16*it..+15)
    const int wave = threadIdx.x >> 5;
    const int lane = threadIdx.x & 31;
    const int m    = lane & 15;
    const int hg   = lane >> 4;

    const size_t fbase = (size_t)bh * NT * ND;

    // A tiles for both branches: rows 16*it + m, reused across all column tiles
    v16h a1[2], a2[2];
    {
        const _Float16* r1 = qf1 + fbase + (size_t)(16 * it + m) * ND;
        const _Float16* r2 = qf2 + fbase + (size_t)(16 * it + m) * ND;
#pragma unroll
        for (int ki = 0; ki < 2; ++ki) {
            const int kb = ki * 32;
            a1[ki] = cat8(*(const v8h*)(r1 + kb + 16 * hg),
                          *(const v8h*)(r1 + kb + 16 * hg + 8));
            a2[ki] = cat8(*(const v8h*)(r2 + kb + 16 * hg),
                          *(const v8h*)(r2 + kb + 16 * hg + 8));
        }
    }

    // causal: only column tiles j <= it+1 can contain kept elements (col <= row+1)
    const int jmax = (it + 1 < 63) ? (it + 1) : 63;
    for (int j = wave; j <= jmax; j += 8) {
        const int cb = j * 16;
        v8f c1 = {}, c2 = {};
        const _Float16* k1r = kf1 + fbase + (size_t)(cb + m) * ND + 16 * hg;
        const _Float16* k2r = kf2 + fbase + (size_t)(cb + m) * ND + 16 * hg;
#pragma unroll
        for (int ki = 0; ki < 2; ++ki) {
            const v16h b1 = *(const v16h*)(k1r + ki * 32);  // 32B contiguous
            const v16h b2 = *(const v16h*)(k2r + ki * 32);
            c1 = __builtin_amdgcn_wmma_f32_16x16x32_f16(false, a1[ki], false, b1,
                                                        (short)0, c1, false, false);
            c2 = __builtin_amdgcn_wmma_f32_16x16x32_f16(false, a2[ki], false, b2,
                                                        (short)0, c2, false, false);
        }
        // mask: keep col <= row+1 (tril diagonal=1); only near-diagonal tiles
        if (j >= it) {
#pragma unroll
            for (int r = 0; r < 8; ++r) {
                const int row = 16 * it + r + 8 * hg;
                const int col = cb + m;
                if (col > row + 1) { c1[r] = 0.f; c2[r] = 0.f; }
            }
        }
#pragma unroll
        for (int r = 0; r < 8; ++r) {
            const int rr = r + 8 * hg;
            aw1[rr * NT + cb + m] = c1[r];
            aw2[rr * NT + cb + m] = c2[r];
        }
    }
    __syncthreads();

    const int ncv_raw = (it + 2) * 16;
    const int ncv = ncv_raw < NT ? ncv_raw : NT; // valid (written) column extent

    // row sums: 8 threads per (branch,row); vector (b128) LDS reads
    {
        const int t   = threadIdx.x;
        const int br  = t >> 7;
        const int row = (t >> 3) & 15;
        const int seg = t & 7;
        const float* aw = br ? aw2 : aw1;
        float s = 0.f;
        const int c0  = seg * 128;
        const int c1e = (c0 + 128 < ncv) ? c0 + 128 : ncv;   // multiple of 16
        for (int c = c0; c < c1e; c += 4) {
            const f4 v = *(const f4*)(aw + row * NT + c);
            s += v.x + v.y + v.z + v.w;
        }
        partial[t] = s;
    }
    __syncthreads();
    if (threadIdx.x < 32) {
        const int br  = threadIdx.x >> 4;
        const int row = threadIdx.x & 15;
        float tot = 0.f;
#pragma unroll
        for (int s = 0; s < 8; ++s) tot += partial[br * 128 + row * 8 + s];
        sinv[threadIdx.x] = 1.0f / tot;
    }
    __syncthreads();

    // fused normalize + differential combine, float4 loads/stores
    const float lam = lambdaPtr[0];
    float* orow = out + ((size_t)bh * NT + 16 * it) * NT;
    for (int g = threadIdx.x; g < 16 * NT / 4; g += blockDim.x) {
        const int row = g >> 8;                  // 256 float4 groups per row
        const int col = (g & 255) * 4;
        f4 v;
        if (col < ncv) {
            const float i1 = sinv[row], i2 = sinv[16 + row];
            const f4 x1 = *(const f4*)(aw1 + row * NT + col);
            const f4 x2 = *(const f4*)(aw2 + row * NT + col);
#pragma unroll
            for (int u = 0; u < 4; ++u)
                v[u] = x1[u] * i1 - lam * x2[u] * i2;
        } else {
            v = (f4){0.f, 0.f, 0.f, 0.f};
        }
        *(f4*)(orow + (size_t)row * NT + col) = v;
    }
}

// ---------------------------------------------------------------------------
extern "C" void kernel_launch(void* const* d_in, const int* in_sizes, int n_in,
                              void* d_out, int out_size, void* d_ws, size_t ws_size,
                              hipStream_t stream) {
    const float* q   = (const float*)d_in[0];
    const float* k   = (const float*)d_in[1];
    const float* W1q = (const float*)d_in[2];
    const float* W1k = (const float*)d_in[3];
    const float* W2q = (const float*)d_in[4];
    const float* W2k = (const float*)d_in[5];
    const float* lq1 = (const float*)d_in[6];
    const float* lk1 = (const float*)d_in[7];
    const float* lq2 = (const float*)d_in[8];
    const float* lk2 = (const float*)d_in[9];
    float* out = (float*)d_out;

    char* ws = (char*)d_ws;
    float* lam = (float*)ws;                           // 64B reserved
    _Float16* feat = (_Float16*)(ws + 64);
    const size_t FE = (size_t)NBH * NT * ND;           // elements per feature buf
    _Float16* qf1 = feat;
    _Float16* kf1 = feat + FE;
    _Float16* qf2 = feat + 2 * FE;
    _Float16* kf2 = feat + 3 * FE;
    _Float16* Wt  = feat + 4 * FE;                     // 4*16*64*64 f16 = 512KB

    wtr_kernel<<<dim3(64), 256, 0, stream>>>(W1q, W1k, W2q, W2k, Wt);
    lambda_kernel<<<1, 64, 0, stream>>>(lq1, lk1, lq2, lk2, lam);
    feat_kernel<<<dim3(NBH, 16, 4), 128, 0, stream>>>(q, k, Wt, qf1, kf1, qf2, kf2);
    const size_t smemBytes = (size_t)(2 * 16 * NT + 256 + 32) * sizeof(float);
    attn_kernel<<<dim3(NBH * 64), 256, smemBytes, stream>>>(qf1, kf1, qf2, kf2, lam, out);
}
